// BinaryMaskBilateralFilter_55791625175591
// MI455X (gfx1250) — matrix operand order repaired and checked by reading, not compiled, not run on hardware
//
#include <hip/hip_runtime.h>
#include <math.h>

typedef __attribute__((ext_vector_type(16))) _Float16 v16h;
typedef __attribute__((ext_vector_type(8)))  _Float16 v8h;
typedef __attribute__((ext_vector_type(2)))  _Float16 v2h;
typedef __attribute__((ext_vector_type(8)))  float    v8f;

#define WPB 4          // waves per block (wave32)
#define IMG 512

// g[d] for d in [0,6] (symmetric 7-tap normalized Gaussian), 0 outside.
__device__ __forceinline__ float gsel(int d, float w0, float w1, float w2, float w3) {
    if (d < 0 || d > 6) return 0.0f;
    int dd = d < 3 ? 3 - d : d - 3;
    return dd == 0 ? w3 : dd == 1 ? w2 : dd == 2 ? w1 : w0;
}

__device__ __forceinline__ int clampi(int v, int lo, int hi) {
    return v < lo ? lo : (v > hi ? hi : v);   // -> v_med3_i32
}

// One bilateral-update iteration:  out = w*blur(m) + (1-w)*m,  w = exp(-10*var)
// var = max(blur(m^2) - blur(m)^2, 0).  Each wave produces one 16x16 tile.
template <int FINAL>
__global__ __launch_bounds__(WPB * 32) void
bmbf_iter_kernel(const float* __restrict__ src, float* __restrict__ dst,
                 float w0, float w1, float w2, float w3)
{
    // Per-wave scratch: [wave][m / m^2][column n][k] (k-major, 80B col stride)
    __shared__ _Float16 sH[WPB][2][16][40];
    // Per-wave zero-padded 7-tap table: sG[24+d] = g[d], d in [0,6], else 0.
    __shared__ _Float16 sG[WPB][64];

    const int lane = threadIdx.x & 31;
    const int wave = threadIdx.x >> 5;
    const int half = lane >> 4;     // lane half (0: lanes 0-15, 1: lanes 16-31)
    const int n    = lane & 15;

    // Tile id is wave-uniform: pin it to an SGPR so the image base pointer is
    // scalar -> SADDR-form loads (s[base] + 32-bit voffset + imm), no 64-bit
    // vector address arithmetic.
    const int tile = __builtin_amdgcn_readfirstlane(blockIdx.x * WPB + wave);
    const int img  = tile >> 10;            // 32x32 tiles per 512x512 image
    const int ty   = (tile >> 5) & 31;
    const int tx   = tile & 31;
    const int y0   = ty * 16;
    const int x0   = tx * 16;

    const float* __restrict__ im = src + (size_t)img * (IMG * IMG);
    float*       __restrict__ om = dst + (size_t)img * (IMG * IMG);

    // Build the tap table (2 stores/lane); same-wave LDS is in-order.
    sG[wave][lane]      = (_Float16)gsel(lane - 24,      w0, w1, w2, w3);
    sG[wave][lane + 32] = (_Float16)gsel(lane + 32 - 24, w0, w1, w2, w3);
    asm volatile("s_wait_dscnt 0" ::: "memory");

    // ---- banded operands from the table ------------------------------
    // Horizontal blur: D = A * Bh, Bh[k][n] = g[k-n].
    // B layout (32x16 f16): element e -> K = e + (e & 8) + (half << 3), N = n.
    const _Float16* gt = &sG[wave][0];
    v16h bh;
    {
        const int base = 24 + (half << 3) - n;      // two contiguous runs
        #pragma unroll
        for (int e = 0; e < 16; ++e) bh[e] = gt[base + e + (e & 8)];
    }
    // Vertical blur: D = Gv * B, Gv[m][k] = g[k-m], m = n.
    // A layout (16x32 f16): element e -> K = e + (e & 8) + (half << 3).
    v16h gv;
    {
        const int b0 = 24 + (half << 3) - n;        // two contiguous runs
        #pragma unroll
        for (int e = 0; e < 8; ++e) {
            gv[e]     = gt[b0 + e];
            gv[e + 8] = gt[b0 + 16 + e];
        }
    }

    // Window spans rows y0-3..y0+18, cols x0-3..x0+28 -> interior iff tx,ty in [1,30].
    // ty/tx are scalar now, so this is a plain scalar branch.
    const bool interior = (unsigned)(ty - 1) <= 29u && (unsigned)(tx - 1) <= 29u;

    // ---- horizontal pass: two 16-row A tiles cover rows y0-3 .. y0+18
    v8f h1 = {}, h1s = {}, h2 = {}, h2s = {};
    #pragma unroll
    for (int t = 0; t < 2; ++t) {
        const int row = y0 - 3 + 6 * t + n;     // A-matrix M = n
        v16h a;
        if (interior) {
            // Straight-line loads: 32-bit offset + immediate offsets per run.
            const int off = row * IMG + (x0 - 3) + (half << 3);
            #pragma unroll
            for (int e = 0; e < 8; ++e) {
                a[e]     = (_Float16)im[off + e];        // K = e      (+half*8)
                a[e + 8] = (_Float16)im[off + e + 16];   // K = e + 16 (+half*8)
            }
        } else {
            // Branchless edge handling: clamp address (always-safe load),
            // then select 0 for out-of-range taps (zero padding).
            const bool rok   = (unsigned)row < (unsigned)IMG;
            const int  rbase = clampi(row, 0, IMG - 1) * IMG;
            #pragma unroll
            for (int e = 0; e < 16; ++e) {
                int K = e + (e & 8) + (half << 3);
                int c = x0 - 3 + K;
                float v = im[rbase + clampi(c, 0, IMG - 1)];
                a[e] = (_Float16)((rok && (unsigned)c < (unsigned)IMG) ? v : 0.0f);
            }
        }
        v16h as = a * a;                        // v_pk_mul_f16, squares for blur(m^2)
        if (t == 0) {
            h1  = __builtin_amdgcn_wmma_f32_16x16x32_f16(false, a,  false, bh, (short)0, h1,  false, false);
            h1s = __builtin_amdgcn_wmma_f32_16x16x32_f16(false, as, false, bh, (short)0, h1s, false, false);
        } else {
            h2  = __builtin_amdgcn_wmma_f32_16x16x32_f16(false, a,  false, bh, (short)0, h2,  false, false);
            h2s = __builtin_amdgcn_wmma_f32_16x16x32_f16(false, as, false, bh, (short)0, h2s, false, false);
        }
    }

    // ---- stage H columns into LDS (f16, packed b32 stores) -------------
    {
        const int kb = half ? 8 : 0;    // D layout: VGPR r holds row r (+8 for upper half)
        #pragma unroll
        for (int i = 0; i < 4; ++i) {   // k offsets 0..15
            v2h p0; p0[0] = (_Float16)h1[2*i];  p0[1] = (_Float16)h1[2*i+1];
            v2h p1; p1[0] = (_Float16)h1s[2*i]; p1[1] = (_Float16)h1s[2*i+1];
            *(v2h*)&sH[wave][0][n][kb + 2*i] = p0;
            *(v2h*)&sH[wave][1][n][kb + 2*i] = p1;
        }
        if (half) {
            #pragma unroll
            for (int i = 1; i < 4; ++i) {   // tile2 rows r=2i..2i+1 -> k=2i+14 (16..21)
                v2h p0; p0[0] = (_Float16)h2[2*i];  p0[1] = (_Float16)h2[2*i+1];
                v2h p1; p1[0] = (_Float16)h2s[2*i]; p1[1] = (_Float16)h2s[2*i+1];
                *(v2h*)&sH[wave][0][n][14 + 2*i] = p0;
                *(v2h*)&sH[wave][1][n][14 + 2*i] = p1;
            }
        } else {
            v2h z; z[0] = (_Float16)0.0f; z[1] = (_Float16)0.0f;
            #pragma unroll
            for (int i = 0; i < 5; ++i) {   // zero pad k=22..31 (0*NaN = NaN in WMMA!)
                *(v2h*)&sH[wave][0][n][22 + 2*i] = z;
                *(v2h*)&sH[wave][1][n][22 + 2*i] = z;
            }
        }
    }
    __syncthreads();

    // ---- vertical pass: B element e -> k = e + (e & 8) + (half ? 8 : 0), col n
    const int kb2 = half ? 8 : 0;
    const v8h* c0 = (const v8h*)&sH[wave][0][n][kb2];
    const v8h* c1 = (const v8h*)&sH[wave][1][n][kb2];
    v8h lo0 = c0[0], hi0 = c0[2];
    v8h lo1 = c1[0], hi1 = c1[2];
    v16h bv  = __builtin_shufflevector(lo0, hi0, 0,1,2,3,4,5,6,7,8,9,10,11,12,13,14,15);
    v16h bvs = __builtin_shufflevector(lo1, hi1, 0,1,2,3,4,5,6,7,8,9,10,11,12,13,14,15);

    v8f fmean = {}, msq = {};
    fmean = __builtin_amdgcn_wmma_f32_16x16x32_f16(false, gv, false, bv,  (short)0, fmean, false, false);
    msq   = __builtin_amdgcn_wmma_f32_16x16x32_f16(false, gv, false, bvs, (short)0, msq,   false, false);

    // ---- elementwise epilogue (D layout: VGPR r -> row r(+8), col n) ---
    const int pbase = (y0 + (half << 3)) * IMG + x0 + n;
    #pragma unroll
    for (int r = 0; r < 8; ++r) {
        float mean = fmean[r];
        float var  = fmaxf(msq[r] - mean * mean, 0.0f);
        float w    = __expf(-10.0f * var);
        float prev = im[pbase + r * IMG];
        float out  = prev + w * (mean - prev);
        if (FINAL) out = out > 0.5f ? 1.0f : 0.0f;
        om[pbase + r * IMG] = out;
    }
}

extern "C" void kernel_launch(void* const* d_in, const int* in_sizes, int n_in,
                              void* d_out, int out_size, void* d_ws, size_t ws_size,
                              hipStream_t stream)
{
    const float* x   = (const float*)d_in[0];
    float*       out = (float*)d_out;
    float*       ws  = (float*)d_ws;

    // Exact normalized separable Gaussian weights (sigma = 1.5, K = 7).
    double g[7], s = 0.0;
    for (int i = 0; i < 7; ++i) {
        double d = i - 3.0;
        g[i] = exp(-(d * d) / (2.0 * 1.5 * 1.5));
        s += g[i];
    }
    float w0 = (float)(g[0] / s), w1 = (float)(g[1] / s);
    float w2 = (float)(g[2] / s), w3 = (float)(g[3] / s);

    int total  = in_sizes[0];                 // B*C*H*W
    int images = total / (IMG * IMG);         // 64
    int tiles  = images * (IMG / 16) * (IMG / 16);
    dim3 block(WPB * 32);
    dim3 grid(tiles / WPB);

    // iter 1: x -> ws ; iter 2 (+ threshold): ws -> out
    bmbf_iter_kernel<0><<<grid, block, 0, stream>>>(x,  ws,  w0, w1, w2, w3);
    bmbf_iter_kernel<1><<<grid, block, 0, stream>>>(ws, out, w0, w1, w2, w3);
}
